// HRAInjectedLinear_40303973106330
// MI455X (gfx1250) — matrix operand order repaired and checked by reading, not compiled, not run on hardware
//
#include <hip/hip_runtime.h>
#include <hip/hip_bf16.h>

// ---------------------------------------------------------------------------
// HRA-injected linear for MI455X (gfx1250, wave32, WMMA).
//   W' = W (I - U T U^T)  (compact-WY of 8 Householder reflections)
//   out = x @ W'^T        (bf16 WMMA, f32 accumulate, async global->LDS)
// ---------------------------------------------------------------------------

typedef __attribute__((ext_vector_type(8)))  float  v8f;
typedef __attribute__((ext_vector_type(8)))  __bf16 v8bf;
typedef __attribute__((ext_vector_type(16))) __bf16 v16bf;

#define D_IN   4096
#define D_OUT  4096
#define RR     8
#define M_TOT  8192            // B*S = 4*2048

static __device__ inline __bf16 f2bf(float f) {
  union { float f; unsigned u; } v; v.f = f;
  unsigned r = v.u + 0x7FFFu + ((v.u >> 16) & 1u);   // round-to-nearest-even
  union { unsigned short s; __bf16 b; } o;
  o.s = (unsigned short)(r >> 16);
  return o.b;
}

// ---------------------------------------------------------------------------
// Kernel 1: normalize U columns, Gram matrix, build 8x8 T (compact WY).
// ---------------------------------------------------------------------------
__global__ __launch_bounds__(256) void hra_prep_kernel(
    const float* __restrict__ hra_u,   // [D_IN][RR] row-major
    float* __restrict__ u_out,         // [D_IN][RR] normalized
    float* __restrict__ T_out)         // [RR][RR]
{
  __shared__ float red[256];
  __shared__ float G[RR][RR];
  __shared__ float sh_rinv[RR];
  const int t = threadIdx.x;

  float g[36];
  #pragma unroll
  for (int e = 0; e < 36; ++e) g[e] = 0.f;

  for (int k = t; k < D_IN; k += 256) {
    const float4* p = (const float4*)(hra_u + (size_t)k * RR);
    float4 a = p[0], b = p[1];
    float row[RR] = {a.x, a.y, a.z, a.w, b.x, b.y, b.z, b.w};
    int e = 0;
    #pragma unroll
    for (int i = 0; i < RR; ++i)
      #pragma unroll
      for (int j = i; j < RR; ++j)
        g[e++] += row[i] * row[j];
  }

  {
    int e = 0;
    for (int i = 0; i < RR; ++i)
      for (int j = i; j < RR; ++j) {
        red[t] = g[e];
        __syncthreads();
        for (int s = 128; s > 0; s >>= 1) {
          if (t < s) red[t] += red[t + s];
          __syncthreads();
        }
        if (t == 0) G[i][j] = red[0];
        __syncthreads();
        ++e;
      }
  }

  if (t == 0) {
    float rinv[RR];
    for (int i = 0; i < RR; ++i) rinv[i] = rsqrtf(G[i][i]);
    float Gn[RR][RR];
    for (int i = 0; i < RR; ++i)
      for (int j = i; j < RR; ++j)
        Gn[i][j] = G[i][j] * rinv[i] * rinv[j];
    float T[RR][RR];
    for (int i = 0; i < RR; ++i)
      for (int j = 0; j < RR; ++j) T[i][j] = 0.f;
    for (int j = 0; j < RR; ++j) {
      T[j][j] = 2.f;
      for (int i = 0; i < j; ++i) {
        float s = 0.f;
        for (int m = i; m < j; ++m) s += T[i][m] * Gn[m][j];
        T[i][j] = -2.f * s;
      }
    }
    for (int i = 0; i < RR; ++i)
      for (int j = 0; j < RR; ++j) T_out[i * RR + j] = T[i][j];
    for (int i = 0; i < RR; ++i) sh_rinv[i] = rinv[i];
  }
  __syncthreads();

  for (int k = t; k < D_IN; k += 256) {
    const float4* p = (const float4*)(hra_u + (size_t)k * RR);
    float4 a = p[0], b = p[1];
    float4* q = (float4*)(u_out + (size_t)k * RR);
    q[0] = make_float4(a.x * sh_rinv[0], a.y * sh_rinv[1], a.z * sh_rinv[2], a.w * sh_rinv[3]);
    q[1] = make_float4(b.x * sh_rinv[4], b.y * sh_rinv[5], b.z * sh_rinv[6], b.w * sh_rinv[7]);
  }
}

// ---------------------------------------------------------------------------
// Kernel 2: Y = W @ U (one wave per output row), then Z = Y @ T.
// ---------------------------------------------------------------------------
__global__ __launch_bounds__(256) void wu_z_kernel(
    const float* __restrict__ W,   // [D_OUT][D_IN]
    const float* __restrict__ u,   // [D_IN][RR]
    const float* __restrict__ T,   // [RR][RR]
    float* __restrict__ Z)         // [D_OUT][RR]
{
  const int wave = threadIdx.x >> 5;
  const int lane = threadIdx.x & 31;
  const int n = blockIdx.x * 8 + wave;
  const float* wrow = W + (size_t)n * D_IN;

  float acc[RR];
  #pragma unroll
  for (int j = 0; j < RR; ++j) acc[j] = 0.f;

  for (int k = lane; k < D_IN; k += 32) {
    float wv = wrow[k];
    const float4* p = (const float4*)(u + (size_t)k * RR);
    float4 a = p[0], b = p[1];
    acc[0] += wv * a.x; acc[1] += wv * a.y; acc[2] += wv * a.z; acc[3] += wv * a.w;
    acc[4] += wv * b.x; acc[5] += wv * b.y; acc[6] += wv * b.z; acc[7] += wv * b.w;
  }
  #pragma unroll
  for (int j = 0; j < RR; ++j) {
    float v = acc[j];
    #pragma unroll
    for (int s = 16; s > 0; s >>= 1) v += __shfl_down(v, s, 32);
    acc[j] = v;
  }
  if (lane == 0) {
    float z[RR];
    #pragma unroll
    for (int j = 0; j < RR; ++j) {
      float s = 0.f;
      for (int i = 0; i <= j; ++i) s += acc[i] * T[i * RR + j];
      z[j] = s;
    }
    float4* q = (float4*)(Z + (size_t)n * RR);
    q[0] = make_float4(z[0], z[1], z[2], z[3]);
    q[1] = make_float4(z[4], z[5], z[6], z[7]);
  }
}

// ---------------------------------------------------------------------------
// Kernel 3: Wb[n][k] = bf16( W[n][k] - dot(Z[n][:], u[k][:]) )
// ---------------------------------------------------------------------------
__global__ __launch_bounds__(256) void build_wb_kernel(
    const float* __restrict__ W,
    const float* __restrict__ u,
    const float* __restrict__ Z,
    __bf16* __restrict__ Wb)
{
  const size_t idx = (size_t)blockIdx.x * 256 + threadIdx.x;  // 8 elems each
  const int n  = (int)(idx >> 9);
  const int k0 = (int)(idx & 511) << 3;
  const float4* zp = (const float4*)(Z + (size_t)n * RR);
  float4 z0 = zp[0], z1 = zp[1];
  const float* wrow = W + (size_t)n * D_IN + k0;

  v8bf outv;
  #pragma unroll
  for (int e = 0; e < 8; ++e) {
    const float4* up = (const float4*)(u + (size_t)(k0 + e) * RR);
    float4 a = up[0], b = up[1];
    float d = z0.x * a.x + z0.y * a.y + z0.z * a.z + z0.w * a.w
            + z1.x * b.x + z1.y * b.y + z1.z * b.z + z1.w * b.w;
    outv[e] = f2bf(wrow[e] - d);
  }
  *(v8bf*)(Wb + (size_t)n * D_IN + k0) = outv;
}

// ---------------------------------------------------------------------------
// Kernel 4: x -> bf16
// ---------------------------------------------------------------------------
__global__ __launch_bounds__(256) void cvt_x_kernel(
    const float* __restrict__ x, __bf16* __restrict__ xb)
{
  const size_t idx = ((size_t)blockIdx.x * 256 + threadIdx.x) * 8;
  const float4* p = (const float4*)(x + idx);
  float4 a = p[0], b = p[1];
  v8bf vv;
  vv[0] = f2bf(a.x); vv[1] = f2bf(a.y); vv[2] = f2bf(a.z); vv[3] = f2bf(a.w);
  vv[4] = f2bf(b.x); vv[5] = f2bf(b.y); vv[6] = f2bf(b.z); vv[7] = f2bf(b.w);
  *(v8bf*)(xb + idx) = vv;
}

// ---------------------------------------------------------------------------
// Kernel 5: C[m][n] = sum_k A[m][k] * B[n][k]   (bf16 WMMA, f32 accum)
// Block tile 128x256, 8 waves of 64x64 (16 WMMA / wave / slab).
// K staged in 32-wide slabs, double-buffered LDS filled with
// GLOBAL_LOAD_ASYNC_TO_LDS_B128 (ASYNCcnt), padded stride 40 bf16.
// ---------------------------------------------------------------------------
#define BM 128
#define BN 256
#define BK 32
#define LDSS 40   // padded bf16 row stride (80B) -> conflict-free ds_load_b128

__global__ __launch_bounds__(256) void gemm_wmma_kernel(
    const __bf16* __restrict__ A,   // [M_TOT][D_IN]
    const __bf16* __restrict__ B,   // [D_OUT][D_IN]
    float* __restrict__ C)          // [M_TOT][D_OUT]
{
  __shared__ __bf16 As[2][BM][LDSS];   // 20 KB
  __shared__ __bf16 Bs[2][BN][LDSS];   // 40 KB

  const int tid    = threadIdx.x;
  const int lane   = tid & 31;
  const int waveId = tid >> 5;
  const int wm = (waveId & 1) * 64;    // wave M offset (2 waves in M)
  const int wn = (waveId >> 1) * 64;   // wave N offset (4 waves in N)
  const int m0 = blockIdx.y * BM;
  const int n0 = blockIdx.x * BN;

  // async staging roles: A -> thread covers (row=tid>>1, 16 cols at (tid&1)*16)
  //                      B -> thread covers full 32-col row tid
  const int arow = tid >> 1;
  const int acol = (tid & 1) * 16;
  const int brow = tid;

  // raw LDS byte offsets (generic LDS addresses: low 32 bits = LDS offset)
  const unsigned ldsA = (unsigned)(size_t)(&As[0][0][0]);
  const unsigned ldsB = (unsigned)(size_t)(&Bs[0][0][0]);
  const unsigned ABUF = (unsigned)(BM * LDSS * 2);
  const unsigned BBUF = (unsigned)(BN * LDSS * 2);
  const unsigned aoff0 = ldsA + ((unsigned)arow * LDSS + (unsigned)acol) * 2u;
  const unsigned boff0 = ldsB + ((unsigned)brow * LDSS) * 2u;

  const __bf16* agp = A + (size_t)(m0 + arow) * D_IN + acol;
  const __bf16* bgp = B + (size_t)(n0 + brow) * D_IN;

  // fragment addressing per ISA 16-bit layouts (lane<16 = K low half)
  const int laneM = lane & 15;
  const int kselA = (lane < 16) ? 0 : 8;    // A: K 0..7 / 8..15 (+16 for hi)
  const int kselB = (lane < 16) ? 0 : 8;    // B: K 0..7 / 8..15 (+16 for hi)

  // issue 6 async b128 loads (2 for A slab, 4 for B slab) into buffer `buf`
  auto issue_slab = [&](int kt, int buf) {
    const __bf16* ga = agp + kt;
    const __bf16* gb = bgp + kt;
    unsigned la = aoff0 + (unsigned)buf * ABUF;
    unsigned lb = boff0 + (unsigned)buf * BBUF;
    asm volatile("global_load_async_to_lds_b128 %0, %1, off"
                 :: "v"(la),        "v"(ga)      : "memory");
    asm volatile("global_load_async_to_lds_b128 %0, %1, off"
                 :: "v"(la + 16u),  "v"(ga + 8)  : "memory");
    asm volatile("global_load_async_to_lds_b128 %0, %1, off"
                 :: "v"(lb),        "v"(gb)      : "memory");
    asm volatile("global_load_async_to_lds_b128 %0, %1, off"
                 :: "v"(lb + 16u),  "v"(gb + 8)  : "memory");
    asm volatile("global_load_async_to_lds_b128 %0, %1, off"
                 :: "v"(lb + 32u),  "v"(gb + 16) : "memory");
    asm volatile("global_load_async_to_lds_b128 %0, %1, off"
                 :: "v"(lb + 48u),  "v"(gb + 24) : "memory");
  };

  v8f acc[4][4];
  #pragma unroll
  for (int i = 0; i < 4; ++i)
    #pragma unroll
    for (int j = 0; j < 4; ++j)
      acc[i][j] = (v8f){0.f, 0.f, 0.f, 0.f, 0.f, 0.f, 0.f, 0.f};

  issue_slab(0, 0);

  for (int kt = 0, it = 0; kt < D_IN; kt += BK, ++it) {
    const int buf = it & 1;
    if (kt + BK < D_IN) {
      issue_slab(kt + BK, buf ^ 1);
      // 6 newer loads may be in flight; in-order completion => slab `it` done
      asm volatile("s_wait_asynccnt 0x6" ::: "memory");
    } else {
      asm volatile("s_wait_asynccnt 0x0" ::: "memory");
    }
    __syncthreads();   // slab visible to all waves

    v16bf af[4], bfr[4];
    #pragma unroll
    for (int i = 0; i < 4; ++i) {
      const __bf16* pa = &As[buf][wm + i * 16 + laneM][0];
      v8bf lo = *(const v8bf*)(pa + kselA);
      v8bf hi = *(const v8bf*)(pa + kselA + 16);
      #pragma unroll
      for (int e = 0; e < 8; ++e) { af[i][e] = lo[e]; af[i][8 + e] = hi[e]; }
    }
    #pragma unroll
    for (int j = 0; j < 4; ++j) {
      const __bf16* pb = &Bs[buf][wn + j * 16 + laneM][0];
      v8bf lo = *(const v8bf*)(pb + kselB);
      v8bf hi = *(const v8bf*)(pb + kselB + 16);
      #pragma unroll
      for (int e = 0; e < 8; ++e) { bfr[j][e] = lo[e]; bfr[j][8 + e] = hi[e]; }
    }
    #pragma unroll
    for (int i = 0; i < 4; ++i)
      #pragma unroll
      for (int j = 0; j < 4; ++j)
        acc[i][j] = __builtin_amdgcn_wmma_f32_16x16x32_bf16(
            false, af[i], false, bfr[j], (short)0, acc[i][j], false, false);

    __syncthreads();   // all waves done with this buffer before refill
  }

  // epilogue: VGPR r -> M=r (lanes 0-15) / M=r+8 (lanes 16-31); N = lane&15
  const int rowAdd = (lane < 16) ? 0 : 8;
  #pragma unroll
  for (int i = 0; i < 4; ++i)
    #pragma unroll
    for (int j = 0; j < 4; ++j) {
      const int col = n0 + wn + j * 16 + laneM;
      #pragma unroll
      for (int r = 0; r < 8; ++r) {
        const int row = m0 + wm + i * 16 + rowAdd + r;
        C[(size_t)row * D_OUT + col] = acc[i][j][r];
      }
    }
}

// ---------------------------------------------------------------------------
// Host launcher
// ---------------------------------------------------------------------------
extern "C" void kernel_launch(void* const* d_in, const int* in_sizes, int n_in,
                              void* d_out, int out_size, void* d_ws, size_t ws_size,
                              hipStream_t stream) {
  (void)in_sizes; (void)n_in; (void)out_size; (void)ws_size;
  const float* x     = (const float*)d_in[0];   // [4,2048,4096]
  const float* W     = (const float*)d_in[1];   // [4096,4096]
  const float* hra_u = (const float*)d_in[2];   // [4096,8]
  float* out = (float*)d_out;                   // [8192,4096]

  char* ws = (char*)d_ws;
  float*  u  = (float*)(ws);                              // 128 KiB
  float*  T  = (float*)(ws + 131072);                     // 1 KiB slot
  float*  Z  = (float*)(ws + 132096);                     // 128 KiB
  __bf16* Wb = (__bf16*)(ws + 263168);                    // 32 MiB
  __bf16* xb = (__bf16*)(ws + 33817600);                  // 64 MiB

  hipLaunchKernelGGL(hra_prep_kernel, dim3(1), dim3(256), 0, stream, hra_u, u, T);
  hipLaunchKernelGGL(wu_z_kernel, dim3(D_OUT / 8), dim3(256), 0, stream, W, u, T, Z);
  hipLaunchKernelGGL(build_wb_kernel, dim3((D_OUT * (size_t)D_IN) / (256 * 8)), dim3(256),
                     0, stream, W, u, Z, Wb);
  hipLaunchKernelGGL(cvt_x_kernel, dim3(((size_t)M_TOT * D_IN) / (256 * 8)), dim3(256),
                     0, stream, x, xb);
  hipLaunchKernelGGL(gemm_wmma_kernel, dim3(D_OUT / BN, M_TOT / BM), dim3(256),
                     0, stream, xb, Wb, out);
}
